// NonLocal_21646635171900
// MI455X (gfx1250) — matrix operand order repaired and checked by reading, not compiled, not run on hardware
//
#include <hip/hip_runtime.h>
#include <hip/hip_bf16.h>

// ---------------- types ----------------
typedef __attribute__((ext_vector_type(16))) __bf16          v16bf;
typedef __attribute__((ext_vector_type(8)))  float           v8f;
typedef __attribute__((ext_vector_type(4)))  float           v4f;
typedef __attribute__((ext_vector_type(8)))  unsigned short  ushort8;

// problem constants
#define BB   16
#define C    256
#define IC   128
#define HW   4096
#define NN   2048   // attention N = HW/2 (reshape quirk)

#define LDSPITCH 40  // halfs per 32-K row slice (80B, multiple of 16B, conflict-free)

#if defined(__AMDGCN__)
#define ASYNC_COPY 1
#else
#define ASYNC_COPY 0
#endif

// ---------------- helpers ----------------
__device__ inline unsigned short f2bf(float f) {
    unsigned int b = __builtin_bit_cast(unsigned int, f);
    b += 0x7FFFu + ((b >> 16) & 1u);           // round-to-nearest-even
    return (unsigned short)(b >> 16);
}

__device__ inline v8f wmma_bf16(v16bf a, v16bf b, v8f c) {
    return __builtin_amdgcn_wmma_f32_16x16x32_bf16(false, a, false, b,
                                                   (short)0, c, false, false);
}

// ---- global-direct fragment loaders (K-contiguous layouts) ----
__device__ inline v16bf load_frag_a(const unsigned short* __restrict__ A,
                                    int ld, int row0, int k0) {
    int l = threadIdx.x & 31;
    int r = l & 15, hi = l >> 4;
    const unsigned short* p = A + (size_t)(row0 + r) * ld + k0 + hi * 8;
    union { ushort8 h[2]; v16bf v; } u;
    u.h[0] = *(const ushort8*)(p);
    u.h[1] = *(const ushort8*)(p + 16);
    return u.v;
}

__device__ inline v16bf load_frag_b(const unsigned short* __restrict__ Bt,
                                    int ld, int k0, int n0) {
    int l = threadIdx.x & 31;
    int cn = l & 15, hi = l >> 4;
    const ushort8* q = (const ushort8*)(Bt + (size_t)(n0 + cn) * ld + k0 + hi * 8);
    union { ushort8 h[2]; v16bf v; } u;
    u.h[0] = q[0];
    u.h[1] = q[2];
    return u.v;
}

// ---- LDS fragment loaders (tile slice with pitch LDSPITCH) ----
__device__ inline v16bf lds_frag_a(const unsigned short* base, int row0) {
    int l = threadIdx.x & 31;
    int r = l & 15, hi = l >> 4;
    const unsigned short* p = base + (row0 + r) * LDSPITCH + hi * 8;
    union { ushort8 h[2]; v16bf v; } u;
    u.h[0] = *(const ushort8*)(p);
    u.h[1] = *(const ushort8*)(p + 16);
    return u.v;
}

__device__ inline v16bf lds_frag_b(const unsigned short* base, int col0) {
    int l = threadIdx.x & 31;
    int cn = l & 15, hi = l >> 4;
    const unsigned short* p = base + (col0 + cn) * LDSPITCH + hi * 8;
    union { ushort8 h[2]; v16bf v; } u;
    u.h[0] = *(const ushort8*)(p);
    u.h[1] = *(const ushort8*)(p + 16);
    return u.v;
}

// ---- async global->LDS copy (32B per thread per matrix) ----
// LDS byte offset = low 32 bits of flat address.
__device__ inline void stage32B(const unsigned short* g, unsigned short* s) {
#if ASYNC_COPY
    unsigned lo = (unsigned)(uintptr_t)s;
    unsigned long long ga = (unsigned long long)(uintptr_t)g;
    asm volatile("global_load_async_to_lds_b128 %0, %1, off"
                 :: "v"(lo), "v"(ga) : "memory");
    asm volatile("global_load_async_to_lds_b128 %0, %1, off offset:16"
                 :: "v"(lo), "v"(ga) : "memory");
#else
    ushort8 a0 = ((const ushort8*)g)[0];
    ushort8 a1 = ((const ushort8*)g)[1];
    ((ushort8*)s)[0] = a0;
    ((ushort8*)s)[1] = a1;
#endif
}

__device__ inline void wait_async_all() {
#if ASYNC_COPY
    asm volatile("s_wait_asynccnt 0x0" ::: "memory");
#endif
}

// mask-conv B fragment: column p -> plane (p>>11 ? odd : even), row n=p&2047
__device__ inline v16bf load_frag_b_mask(const unsigned short* __restrict__ outE,
                                         const unsigned short* __restrict__ outO,
                                         size_t boff, int k0, int p_base) {
    int l = threadIdx.x & 31;
    int cn = l & 15, hi = l >> 4;
    int p = p_base + cn;
    const unsigned short* base = ((p >> 11) ? outO : outE) + boff;
    const ushort8* q = (const ushort8*)(base + (size_t)(p & (NN - 1)) * IC + k0 + hi * 8);
    union { ushort8 h[2]; v16bf v; } u;
    u.h[0] = q[0];
    u.h[1] = q[2];
    return u.v;
}

// ---------------- kernels ----------------

// K0: weights -> bf16
__global__ void pack_weights(const float* __restrict__ w_phi,
                             const float* __restrict__ w_theta,
                             const float* __restrict__ w_g,
                             const float* __restrict__ w_mask,
                             unsigned short* __restrict__ Wcat,
                             unsigned short* __restrict__ Wm) {
    int i = blockIdx.x * blockDim.x + threadIdx.x;
    const int n1 = 384 * C;
    if (i < n1) {
        int row = i >> 8, c = i & 255;
        float v = (row < 128) ? w_phi[row * C + c]
                : (row < 256) ? w_theta[(row - 128) * C + c]
                              : w_g[(row - 256) * C + c];
        Wcat[i] = f2bf(v);
    } else if (i < n1 + C * IC) {
        int j = i - n1;
        Wm[j] = f2bf(w_mask[j]);
    }
}

// K1: x[b][c][hw] f32 -> xT[b][hw][c] bf16 (LDS tile transpose)
__global__ void transpose_cast_x(const float* __restrict__ x,
                                 unsigned short* __restrict__ xT) {
    __shared__ float tile[32][33];
    int b = blockIdx.z;
    int hw0 = blockIdx.x * 32, c0 = blockIdx.y * 32;
    const float* xb = x + (size_t)b * C * HW;
#pragma unroll
    for (int j = 0; j < 32; j += 8)
        tile[threadIdx.y + j][threadIdx.x] =
            xb[(size_t)(c0 + threadIdx.y + j) * HW + hw0 + threadIdx.x];
    __syncthreads();
    unsigned short* xtb = xT + (size_t)b * HW * C;
#pragma unroll
    for (int j = 0; j < 32; j += 8)
        xtb[(size_t)(hw0 + threadIdx.y + j) * C + c0 + threadIdx.x] =
            f2bf(tile[threadIdx.x][threadIdx.y + j]);
}

// K2: Y = Wcat(384x256) @ x(256x4096) + bias -> phiT/thetaT transposed, g natural
// K loop fully unrolled (8 steps): scheduler hoists loads, no register copies.
__global__ __launch_bounds__(128) void proj_gemm(
        const unsigned short* __restrict__ Wcat,
        const unsigned short* __restrict__ xT,
        const float* __restrict__ b_phi,
        const float* __restrict__ b_theta,
        const float* __restrict__ b_g,
        unsigned short* __restrict__ phiT,
        unsigned short* __restrict__ thetaT,
        unsigned short* __restrict__ g) {
    int b = blockIdx.z;
    int wave = threadIdx.x >> 5;
    int m0 = blockIdx.x * 64 + wave * 16;      // oc
    int n0 = blockIdx.y * 64;                  // hw
    const unsigned short* Bt = xT + (size_t)b * HW * C;
    v8f acc[4] = {};
#pragma unroll
    for (int k0 = 0; k0 < C; k0 += 32) {
        v16bf a = load_frag_a(Wcat, C, m0, k0);
#pragma unroll
        for (int t = 0; t < 4; ++t) {
            v16bf bf = load_frag_b(Bt, C, k0, n0 + 16 * t);
            acc[t] = wmma_bf16(a, bf, acc[t]);
        }
    }
    int l = threadIdx.x & 31, cn = l & 15, hi = l >> 4;
#pragma unroll
    for (int t = 0; t < 4; ++t) {
        int hw = n0 + 16 * t + cn;
        int m = hw & (NN - 1), ph = hw >> 11;
#pragma unroll
        for (int r = 0; r < 8; ++r) {
            int oc = m0 + r + 8 * hi;
            float v = acc[t][r];
            if (oc < 128) {
                v += b_phi[oc];
                phiT[((size_t)b * NN + m) * C + 2 * oc + ph] = f2bf(v);
            } else if (oc < 256) {
                int o = oc - 128;
                v += b_theta[o];
                thetaT[((size_t)b * NN + m) * C + 2 * o + ph] = f2bf(v);
            } else {
                int o = oc - 256;
                v += b_g[o];
                g[(size_t)b * IC * HW + (size_t)o * HW + hw] = f2bf(v);
            }
        }
    }
}

// K3: S[n][m] = sum_c thetaT[n][c]*phiT[m][c]  (one batch)
// LDS double-buffered 128x128 tile, async global->LDS pipeline, 8 WMMAs/step/wave
__global__ __launch_bounds__(256) void attn_scores(
        const unsigned short* __restrict__ thetaTb,
        const unsigned short* __restrict__ phiTb,
        float* __restrict__ S) {
    __shared__ unsigned short tA[2][128 * LDSPITCH];
    __shared__ unsigned short tB[2][128 * LDSPITCH];
    int tid = threadIdx.x;
    int wave = tid >> 5;                 // 8 waves: 4(M) x 2(N)
    int wm = wave & 3, wn = wave >> 2;
    int bm0 = blockIdx.x * 128;          // n rows
    int bn0 = blockIdx.y * 128;          // m cols
    // copy lane: 256 threads x 32B cover one 128x32 bf16 slice per matrix
    int lrow = tid >> 1;                 // 0..127
    int lchunk = (tid & 1) * 16;         // halfs
    const unsigned short* gA = thetaTb + (size_t)(bm0 + lrow) * C + lchunk;
    const unsigned short* gB = phiTb  + (size_t)(bn0 + lrow) * C + lchunk;
    int soff = lrow * LDSPITCH + lchunk;

    v8f acc[2][4] = {};
    stage32B(gA, &tA[0][soff]);
    stage32B(gB, &tB[0][soff]);
    int buf = 0;
    for (int s = 0; s < 8; ++s) {
        wait_async_all();
        __syncthreads();                 // tile[buf] visible to all waves
        if (s < 7) {                     // copies for next slice fly under WMMAs
            int kn = (s + 1) * 32;
            stage32B(gA + kn, &tA[buf ^ 1][soff]);
            stage32B(gB + kn, &tB[buf ^ 1][soff]);
        }
        v16bf a0 = lds_frag_a(&tA[buf][0], wm * 32);
        v16bf a1 = lds_frag_a(&tA[buf][0], wm * 32 + 16);
#pragma unroll
        for (int t = 0; t < 4; ++t) {
            v16bf bf = lds_frag_b(&tB[buf][0], wn * 64 + 16 * t);
            acc[0][t] = wmma_bf16(a0, bf, acc[0][t]);
            acc[1][t] = wmma_bf16(a1, bf, acc[1][t]);
        }
        __syncthreads();                 // done reading tile[buf] before restage
        buf ^= 1;
    }
    int l = tid & 31, cn = l & 15, hi = l >> 4;
#pragma unroll
    for (int i = 0; i < 2; ++i)
#pragma unroll
        for (int t = 0; t < 4; ++t)
#pragma unroll
            for (int r = 0; r < 8; ++r)
                S[(size_t)(bm0 + wm * 32 + 16 * i + r + 8 * hi) * NN +
                  bn0 + wn * 64 + 16 * t + cn] = acc[i][t][r];
}

// K4a: partial column stats (16 row chunks of 128 -> 32K parallel threads)
__global__ void softmax_stats_part(const float* __restrict__ S,
                                   float* __restrict__ pmax,
                                   float* __restrict__ psum) {
    int col = blockIdx.x * 256 + threadIdx.x;
    int r0 = blockIdx.y * 128;
    float m = -__builtin_inff(), s = 0.f;
    for (int n = r0; n < r0 + 128; ++n) {
        float v = S[(size_t)n * NN + col];
        float nm = fmaxf(m, v);
        s = s * __expf(m - nm) + __expf(v - nm);
        m = nm;
    }
    pmax[(size_t)blockIdx.y * NN + col] = m;
    psum[(size_t)blockIdx.y * NN + col] = s;
}

// K4b: merge partials -> colmax, 1/colsum
__global__ void softmax_stats_merge(const float* __restrict__ pmax,
                                    const float* __restrict__ psum,
                                    float* __restrict__ cmax,
                                    float* __restrict__ crs) {
    int col = blockIdx.x * 256 + threadIdx.x;
    float M = -__builtin_inff();
#pragma unroll
    for (int i = 0; i < 16; ++i) M = fmaxf(M, pmax[(size_t)i * NN + col]);
    float s = 0.f;
#pragma unroll
    for (int i = 0; i < 16; ++i)
        s += psum[(size_t)i * NN + col] * __expf(pmax[(size_t)i * NN + col] - M);
    cmax[col] = M;
    crs[col] = 1.0f / s;
}

// K5: out[n][c] = sum_m softmax(S)[n][m]*g[c][m]; even/odd channel planes.
// True ping-pong (unroll-by-2): slot indices are constants, no register copies.
__global__ __launch_bounds__(128) void attn_pv(
        const float* __restrict__ S,
        const float* __restrict__ cmax,
        const float* __restrict__ crs,
        const unsigned short* __restrict__ gb,
        unsigned short* __restrict__ outE,
        unsigned short* __restrict__ outO, int b) {
    int wave = threadIdx.x >> 5;
    int n0 = blockIdx.x * 64 + wave * 16;      // attention row n
    int c0 = blockIdx.y * 64;                  // channel c
    int l = threadIdx.x & 31;
    int r = l & 15, hi = l >> 4;
    const float* srow = S + (size_t)(n0 + r) * NN;

    v8f acc[4] = {};
    v16bf bfr[2][4];
    v4f sr[2][4];

    auto load_slot = [&](int slot, int k0) {
#pragma unroll
        for (int t = 0; t < 4; ++t) bfr[slot][t] = load_frag_b(gb, NN, k0, c0 + 16 * t);
        int kb0 = k0 + hi * 8, kb1 = k0 + 16 + hi * 8;
        sr[slot][0] = *(const v4f*)(srow + kb0);
        sr[slot][1] = *(const v4f*)(srow + kb0 + 4);
        sr[slot][2] = *(const v4f*)(srow + kb1);
        sr[slot][3] = *(const v4f*)(srow + kb1 + 4);
    };
    auto compute_slot = [&](int slot, int k0) {
        union { unsigned short u[16]; v16bf v; } a;
        int kb0 = k0 + hi * 8, kb1 = k0 + 16 + hi * 8;
#pragma unroll
        for (int j = 0; j < 4; ++j) {
            int kb = ((j < 2) ? kb0 : kb1) + (j & 1) * 4;
#pragma unroll
            for (int i = 0; i < 4; ++i) {
                int k = kb + i;
                a.u[j * 4 + i] = f2bf(__expf(sr[slot][j][i] - cmax[k]) * crs[k]);
            }
        }
#pragma unroll
        for (int t = 0; t < 4; ++t) acc[t] = wmma_bf16(a.v, bfr[slot][t], acc[t]);
    };

    load_slot(0, 0);
    for (int k0 = 0; k0 < NN; k0 += 64) {
        load_slot(1, k0 + 32);           // flies under slot-0 exp+WMMA work
        compute_slot(0, k0);
        if (k0 + 64 < NN) load_slot(0, k0 + 64);
        compute_slot(1, k0 + 32);
    }

    int cn = l & 15;
#pragma unroll
    for (int t = 0; t < 4; ++t) {
        int c = c0 + 16 * t + cn;
        unsigned short* dst = (c & 1) ? outO : outE;
#pragma unroll
        for (int rr = 0; rr < 8; ++rr) {
            int n = n0 + rr + 8 * hi;
            dst[((size_t)b * NN + n) * IC + (c >> 1)] = f2bf(acc[t][rr]);
        }
    }
}

// K6: mask = w_mask(256x128) @ out_view(128x4096) + b_mask + x -> d_out
// K loop fully unrolled (4 steps).
__global__ __launch_bounds__(256) void mask_conv(
        const unsigned short* __restrict__ Wm,
        const unsigned short* __restrict__ outE,
        const unsigned short* __restrict__ outO,
        const float* __restrict__ b_mask,
        const float* __restrict__ x,
        float* __restrict__ out) {
    int b = blockIdx.z;
    int wave = threadIdx.x >> 5;               // 8 waves: 4(M) x 2(N)
    int wm = wave & 3, wn = wave >> 2;
    int m0 = blockIdx.x * 128 + wm * 32;       // oc
    int p0 = blockIdx.y * 128 + wn * 64;       // spatial p
    size_t boff = (size_t)b * NN * IC;
    v8f acc[2][4] = {};
#pragma unroll
    for (int k0 = 0; k0 < IC; k0 += 32) {
        v16bf a0 = load_frag_a(Wm, IC, m0, k0);
        v16bf a1 = load_frag_a(Wm, IC, m0 + 16, k0);
#pragma unroll
        for (int t = 0; t < 4; ++t) {
            v16bf bf = load_frag_b_mask(outE, outO, boff, k0, p0 + 16 * t);
            acc[0][t] = wmma_bf16(a0, bf, acc[0][t]);
            acc[1][t] = wmma_bf16(a1, bf, acc[1][t]);
        }
    }
    int l = threadIdx.x & 31, cn = l & 15, hi = l >> 4;
#pragma unroll
    for (int i = 0; i < 2; ++i)
#pragma unroll
        for (int t = 0; t < 4; ++t) {
            int p = p0 + 16 * t + cn;
#pragma unroll
            for (int r = 0; r < 8; ++r) {
                int oc = m0 + 16 * i + r + 8 * hi;
                size_t idx = ((size_t)b * C + oc) * HW + p;
                out[idx] = acc[i][t][r] + b_mask[oc] + x[idx];
            }
        }
}

// ---------------- host ----------------
extern "C" void kernel_launch(void* const* d_in, const int* in_sizes, int n_in,
                              void* d_out, int out_size, void* d_ws, size_t ws_size,
                              hipStream_t stream) {
    (void)in_sizes; (void)n_in; (void)out_size; (void)ws_size;
    const float* x       = (const float*)d_in[0];
    const float* w_phi   = (const float*)d_in[1];
    const float* b_phi   = (const float*)d_in[2];
    const float* w_theta = (const float*)d_in[3];
    const float* b_theta = (const float*)d_in[4];
    const float* w_g     = (const float*)d_in[5];
    const float* b_g     = (const float*)d_in[6];
    const float* w_mask  = (const float*)d_in[7];
    const float* b_mask  = (const float*)d_in[8];
    float* outp = (float*)d_out;

    char* ws = (char*)d_ws;
    size_t off = 0;
    auto carve = [&](size_t bytes) { size_t o = off; off += (bytes + 255) & ~(size_t)255; return o; };
    unsigned short* Wcat   = (unsigned short*)(ws + carve((size_t)384 * C * 2));
    unsigned short* Wm     = (unsigned short*)(ws + carve((size_t)C * IC * 2));
    unsigned short* xT     = (unsigned short*)(ws + carve((size_t)BB * HW * C * 2));
    unsigned short* phiT   = (unsigned short*)(ws + carve((size_t)BB * NN * C * 2));
    unsigned short* thetaT = (unsigned short*)(ws + carve((size_t)BB * NN * C * 2));
    unsigned short* gbuf   = (unsigned short*)(ws + carve((size_t)BB * IC * HW * 2));
    unsigned short* outE   = (unsigned short*)(ws + carve((size_t)BB * NN * IC * 2));
    unsigned short* outO   = (unsigned short*)(ws + carve((size_t)BB * NN * IC * 2));
    float* cmax            = (float*)(ws + carve((size_t)NN * 4));
    float* crs             = (float*)(ws + carve((size_t)NN * 4));
    float* pmax            = (float*)(ws + carve((size_t)16 * NN * 4));
    float* psum            = (float*)(ws + carve((size_t)16 * NN * 4));
    float* S               = (float*)(ws + carve((size_t)NN * NN * 4)); // per-batch, L2-resident

    pack_weights<<<512, 256, 0, stream>>>(w_phi, w_theta, w_g, w_mask, Wcat, Wm);
    transpose_cast_x<<<dim3(HW / 32, C / 32, BB), dim3(32, 8), 0, stream>>>(x, xT);
    proj_gemm<<<dim3(384 / 64, HW / 64, BB), 128, 0, stream>>>(
        Wcat, xT, b_phi, b_theta, b_g, phiT, thetaT, gbuf);

    for (int b = 0; b < BB; ++b) {
        const unsigned short* thetaTb = thetaT + (size_t)b * NN * C;
        const unsigned short* phiTb   = phiT   + (size_t)b * NN * C;
        const unsigned short* gb      = gbuf   + (size_t)b * IC * HW;
        attn_scores<<<dim3(NN / 128, NN / 128), 256, 0, stream>>>(thetaTb, phiTb, S);
        softmax_stats_part<<<dim3(NN / 256, 16), 256, 0, stream>>>(S, pmax, psum);
        softmax_stats_merge<<<NN / 256, 256, 0, stream>>>(pmax, psum, cmax, crs);
        attn_pv<<<dim3(NN / 64, C / 64), 128, 0, stream>>>(S, cmax, crs, gb, outE, outO, b);
    }

    mask_conv<<<dim3(C / 128, HW / 128, BB), 256, 0, stream>>>(
        Wm, outE, outO, b_mask, x, outp);
}
